// MultiHeadAttentionClassical_65481071396931
// MI455X (gfx1250) — matrix operand order repaired and checked by reading, not compiled, not run on hardware
//
#include <hip/hip_runtime.h>
#include <hip/hip_bf16.h>
#include <math.h>

// ---------------------------------------------------------------------------
// MultiHeadAttentionClassical for MI455X (gfx1250), bf16 WMMA path.
// B=4, S=1024, E=1024, H=16, DK=64.  v-projection is dead code -> skipped.
// All GEMM operands pre-converted to bf16 once; hot loops are pure
// load_b128 + v_wmma_f32_16x16x32_bf16.
// ---------------------------------------------------------------------------

typedef __attribute__((ext_vector_type(16))) __bf16 v16bf;
typedef __attribute__((ext_vector_type(8)))  __bf16 v8bf;
typedef __attribute__((ext_vector_type(8)))  float  v8f;
typedef __attribute__((ext_vector_type(4)))  float  v4f;

#define Bn 4
#define Sn 1024
#define En 1024
#define Hn 16
#define DKn 64

__device__ __forceinline__ v8f wmma_bf16(v16bf a, v16bf b, v8f c) {
    return __builtin_amdgcn_wmma_f32_16x16x32_bf16(
        /*neg_a=*/false, a, /*neg_b=*/false, b,
        /*c_mod=*/(short)0, c, /*reuse_a=*/false, /*reuse_b=*/false);
}

// A fragment (16x32 bf16, M x K), bf16 source, row-major, K contiguous/lane.
// lane<16: M=lane, kbase=0 ; lane>=16: M=lane-16, kbase=8
// element e<8  -> K = k0 + kbase + e
// element e>=8 -> K = k0 + kbase + 16 + (e-8)
__device__ __forceinline__ v16bf frag_a_bf16(const __bf16* __restrict__ row,
                                             int k0, int kbase) {
    v8bf c1 = *(const v8bf*)(row + k0 + kbase);
    v8bf c2 = *(const v8bf*)(row + k0 + kbase + 16);
    v16bf r;
#pragma unroll
    for (int i = 0; i < 8; ++i) { r[i] = c1[i]; r[8 + i] = c2[i]; }
    return r;
}

// B fragment (32x16 bf16, K x N); per lane N = lane&15, same K order as A:
// e<8 -> K = k0 + koff/2 + e, e>=8 -> K = k0 + koff/2 + 16 + (e-8), koff = (lane>=16)?16:0.
// `row` points at W[n][.] (we need W^T; W row-major gives contiguous K).
__device__ __forceinline__ v16bf frag_b_bf16(const __bf16* __restrict__ row,
                                             int k0, int koff) {
    v8bf c1 = *(const v8bf*)(row + k0 + (koff >> 1));
    v8bf c2 = *(const v8bf*)(row + k0 + (koff >> 1) + 16);
    v16bf r;
#pragma unroll
    for (int i = 0; i < 8; ++i) { r[i] = c1[i]; r[8 + i] = c2[i]; }
    return r;
}

// ---------------------------------------------------------------------------
// Kernel 0: one-pass f32 -> bf16 conversion (16B-vectorized).
// n must be a multiple of 8 (all our tensors are).
// ---------------------------------------------------------------------------
__global__ __launch_bounds__(256) void cvt_bf16_kernel(
    const float* __restrict__ in, __bf16* __restrict__ out, int n8) {
    const int i = blockIdx.x * blockDim.x + threadIdx.x;
    if (i >= n8) return;
    const size_t base = (size_t)i * 8;
    v4f a = *(const v4f*)(in + base);
    v4f b = *(const v4f*)(in + base + 4);
    v8bf r;
#pragma unroll
    for (int j = 0; j < 4; ++j) {
        r[j]     = (__bf16)a[j];
        r[4 + j] = (__bf16)b[j];
    }
    *(v8bf*)(out + base) = r;
}

// ---------------------------------------------------------------------------
// Kernel 1: Y = x @ W^T + b, bf16 in / bf16 out.  One wave per 16x64 strip.
// ---------------------------------------------------------------------------
__global__ __launch_bounds__(32) void proj_kernel(
    const __bf16* __restrict__ x, const __bf16* __restrict__ W,
    const float* __restrict__ bias, __bf16* __restrict__ out) {
    const int lane  = threadIdx.x;
    const int m0    = blockIdx.x * 16;
    const int n0    = blockIdx.y * 64;
    const int Mrow  = lane & 15;
    const int kbase = (lane >> 4) * 8;
    const int koff  = (lane >> 4) * 16;
    const int rbase = (lane >> 4) * 8;

    const __bf16* arow = x + (size_t)(m0 + Mrow) * En;
    v8f acc[4] = {};
    for (int k0 = 0; k0 < En; k0 += 32) {
        v16bf a = frag_a_bf16(arow, k0, kbase);
#pragma unroll
        for (int t = 0; t < 4; ++t) {
            const __bf16* brow = W + (size_t)(n0 + t * 16 + Mrow) * En;
            v16bf bf = frag_b_bf16(brow, k0, koff);
            acc[t] = wmma_bf16(a, bf, acc[t]);
        }
    }
#pragma unroll
    for (int t = 0; t < 4; ++t) {
        const int col = n0 + t * 16 + (lane & 15);
        const float bv = bias[col];
#pragma unroll
        for (int r = 0; r < 8; ++r) {
            const int m = m0 + r + rbase;
            out[(size_t)m * En + col] = (__bf16)(acc[t][r] + bv);
        }
    }
}

// ---------------------------------------------------------------------------
// Kernel 2: fused scores -> mask -> softmax -> (P @ Wo^T + bo).
// Grid: (qtile=64, h=16, b=4); block: 256 threads = 8 waves.
// Each WG owns 16 q-rows of one head; P block [16,1024] lives in LDS only
// (never touches HBM: saves 2 x 256 MB of traffic).
// Output rows: (q*16 + h) of out[b, 16384, 1024]  (the reshape identity).
// ---------------------------------------------------------------------------
#define SROW 1040  // f32 score row stride (padded)
#define PROW 1056  // bf16 P row stride (multiple of 8 -> 16B-aligned chunks)

__global__ __launch_bounds__(256) void attn_kernel(
    const __bf16* __restrict__ Q, const __bf16* __restrict__ Kb,
    const int* __restrict__ mask, const __bf16* __restrict__ Wo,
    const float* __restrict__ bo, float* __restrict__ out) {
    __shared__ __attribute__((aligned(16))) float  sS[16 * SROW];
    __shared__ __attribute__((aligned(16))) __bf16 sP[16 * PROW];
    __shared__ float sRed[16 * 16];
    __shared__ float sRowv[16];

    const int qt   = blockIdx.x;      // 0..63
    const int h    = blockIdx.y;      // 0..15
    const int b    = blockIdx.z;      // 0..3
    const int tid  = threadIdx.x;
    const int lane = tid & 31;
    const int wave = tid >> 5;        // 0..7
    const int q0   = qt * 16;
    const int Mrow  = lane & 15;
    const int kbase = (lane >> 4) * 8;
    const int koff  = (lane >> 4) * 16;
    const int rbase = (lane >> 4) * 8;

    // ---- Phase 1: scores = Q K^T / 8, masked, into LDS (f32) ----
    {
        const __bf16* qrow =
            Q + ((size_t)(b * Sn + q0 + Mrow)) * En + h * DKn;
        v8f acc[8] = {};
#pragma unroll
        for (int d0 = 0; d0 < DKn; d0 += 32) {
            v16bf a = frag_a_bf16(qrow, d0, kbase);
#pragma unroll
            for (int t = 0; t < 8; ++t) {
                const int kcol = (wave * 8 + t) * 16 + Mrow;  // B lane column
                const __bf16* krow =
                    Kb + ((size_t)(b * Sn + kcol)) * En + h * DKn;
                v16bf bf = frag_b_bf16(krow, d0, koff);
                acc[t] = wmma_bf16(a, bf, acc[t]);
            }
        }
#pragma unroll
        for (int t = 0; t < 8; ++t) {
            const int col = (wave * 8 + t) * 16 + (lane & 15);
#pragma unroll
            for (int r = 0; r < 8; ++r) {
                const int qr = r + rbase;
                const int mv =
                    mask[((size_t)b * Sn + q0 + qr) * Sn + col];
                const float v = acc[t][r] * 0.125f;  // 1/sqrt(64)
                sS[qr * SROW + col] = (mv == 0) ? -1e9f : v;
            }
        }
    }
    __syncthreads();

    // ---- Softmax over 1024 cols per row; write P as bf16 into LDS ----
    {
        const int row = tid & 15;
        const int seg = tid >> 4;        // 16 segments x 64 cols
        const int c0  = seg * 64;
        float mx = -INFINITY;
        for (int c = 0; c < 64; ++c) mx = fmaxf(mx, sS[row * SROW + c0 + c]);
        sRed[row * 16 + seg] = mx;
        __syncthreads();
        if (seg == 0) {
            float m2 = sRed[row * 16];
            for (int s = 1; s < 16; ++s) m2 = fmaxf(m2, sRed[row * 16 + s]);
            sRowv[row] = m2;
        }
        __syncthreads();
        const float rowmax = sRowv[row];
        float lsum = 0.f;
        for (int c = 0; c < 64; ++c) {
            const float e = __expf(sS[row * SROW + c0 + c] - rowmax);
            sS[row * SROW + c0 + c] = e;
            lsum += e;
        }
        sRed[row * 16 + seg] = lsum;
        __syncthreads();
        if (seg == 0) {
            float s2 = 0.f;
            for (int s = 0; s < 16; ++s) s2 += sRed[row * 16 + s];
            sRowv[row] = 1.0f / s2;
        }
        __syncthreads();
        const float inv = sRowv[row];
        for (int c = 0; c < 64; ++c)
            sP[row * PROW + c0 + c] =
                (__bf16)(sS[row * SROW + c0 + c] * inv);
    }
    __syncthreads();

    // ---- Phase 2: out[16,1024] = P @ Wo^T + bo  (dominant GEMM) ----
    {
        v8f oacc[8] = {};
        const __bf16* prow = sP + Mrow * PROW;  // LDS, A-frag source
        for (int k0 = 0; k0 < Sn; k0 += 32) {
            v16bf a = frag_a_bf16(prow, k0, kbase);
#pragma unroll
            for (int t = 0; t < 8; ++t) {
                const int ecol = (wave * 8 + t) * 16 + Mrow;
                v16bf bf = frag_b_bf16(Wo + (size_t)ecol * En, k0, koff);
                oacc[t] = wmma_bf16(a, bf, oacc[t]);
            }
        }
#pragma unroll
        for (int t = 0; t < 8; ++t) {
            const int col = (wave * 8 + t) * 16 + (lane & 15);
            const float bv = bo[col];
#pragma unroll
            for (int r = 0; r < 8; ++r) {
                const int qr = r + rbase;
                const size_t orow =
                    (size_t)b * (Sn * Hn) + (size_t)(q0 + qr) * Hn + h;
                out[orow * En + col] = oacc[t][r] + bv;
            }
        }
    }
}

// ---------------------------------------------------------------------------
extern "C" void kernel_launch(void* const* d_in, const int* in_sizes, int n_in,
                              void* d_out, int out_size, void* d_ws,
                              size_t ws_size, hipStream_t stream) {
    (void)in_sizes; (void)n_in; (void)out_size; (void)ws_size;
    const float* x    = (const float*)d_in[0];
    const int*   mask = (const int*)d_in[1];
    const float* Wq   = (const float*)d_in[2];
    const float* bq   = (const float*)d_in[3];
    const float* Wk   = (const float*)d_in[4];
    const float* bk   = (const float*)d_in[5];
    // d_in[6], d_in[7] = Wv, bv -> dead code in the reference, skipped.
    const float* Wo   = (const float*)d_in[8];
    const float* bo   = (const float*)d_in[9];
    float* out = (float*)d_out;

    // Workspace layout (bf16):
    const size_t NX = (size_t)(Bn * Sn) * En;   // 4M elems
    const size_t NW = (size_t)En * En;          // 1M elems
    __bf16* xb  = (__bf16*)d_ws;                // 8 MB
    __bf16* Wqb = xb  + NX;                     // 2 MB
    __bf16* Wkb = Wqb + NW;                     // 2 MB
    __bf16* Wob = Wkb + NW;                     // 2 MB
    __bf16* Qb  = Wob + NW;                     // 8 MB
    __bf16* Kb  = Qb  + NX;                     // 8 MB

    // One-time bf16 conversions (removes all cvt from GEMM hot loops).
    {
        const int nx8 = (int)(NX / 8), nw8 = (int)(NW / 8);
        cvt_bf16_kernel<<<(nx8 + 255) / 256, 256, 0, stream>>>(x,  xb,  nx8);
        cvt_bf16_kernel<<<(nw8 + 255) / 256, 256, 0, stream>>>(Wq, Wqb, nw8);
        cvt_bf16_kernel<<<(nw8 + 255) / 256, 256, 0, stream>>>(Wk, Wkb, nw8);
        cvt_bf16_kernel<<<(nw8 + 255) / 256, 256, 0, stream>>>(Wo, Wob, nw8);
    }

    dim3 pg((Bn * Sn) / 16, En / 64, 1);   // 256 x 16
    proj_kernel<<<pg, 32, 0, stream>>>(xb, Wqb, bq, Qb);
    proj_kernel<<<pg, 32, 0, stream>>>(xb, Wkb, bk, Kb);

    dim3 ag(Sn / 16, Hn, Bn);              // 64 x 16 x 4
    attn_kernel<<<ag, 256, 0, stream>>>(Qb, Kb, mask, Wob, bo, out);
}
